// HiMambaBottleneck_39865886441841
// MI455X (gfx1250) — matrix operand order repaired and checked by reading, not compiled, not run on hardware
//
#include <hip/hip_runtime.h>
#include <hip/hip_bf16.h>

#define B_      8
#define C_      384
#define HW_     1024
#define L_      1024
#define NSTATE  16
#define DTRANK  24
#define BL      (B_ * L_)   // 8192

typedef __attribute__((ext_vector_type(16))) __bf16 v16bf;
typedef __attribute__((ext_vector_type(8)))  float  v8f;
typedef __attribute__((ext_vector_type(4)))  unsigned int v4u;
typedef __attribute__((ext_vector_type(8)))  int v8i;
typedef __attribute__((ext_vector_type(4)))  int v4i;
typedef unsigned short u16b;
typedef unsigned int   u32b;

#if defined(__has_builtin)
#if __has_builtin(__builtin_amdgcn_tensor_load_to_lds) && __has_builtin(__builtin_amdgcn_s_wait_tensorcnt)
#define HAVE_TDM 1
#endif
#endif
#ifndef HAVE_TDM
#define HAVE_TDM 0
#endif

__device__ __forceinline__ u16b f2bf(float f) {
    u32b u = __float_as_uint(f);
    u32b r = u + 0x7FFFu + ((u >> 16) & 1u);   // round-to-nearest-even
    return (u16b)(r >> 16);
}
__device__ __forceinline__ float sigm_(float x) { return 1.0f / (1.0f + __expf(-x)); }
__device__ __forceinline__ float silu_(float x) { return x * sigm_(x); }
__device__ __forceinline__ float softplus_(float x) { return (x > 20.0f) ? x : log1pf(__expf(x)); }

// ---- pack f32 weights [rows_src][K] -> bf16 [rows_dst][K], zero-padded rows ----
__global__ void k_pack_w(const float* __restrict__ src, u16b* __restrict__ dst,
                         int rows_src, int rows_dst, int K) {
    int i = blockIdx.x * blockDim.x + threadIdx.x;
    if (i >= rows_dst * K) return;
    int r = i / K;
    dst[i] = (r < rows_src) ? f2bf(src[i]) : (u16b)0;
}

// ---- RMSNorm: NCHW f32 -> channels-last [BL][C] f32 ----
__global__ void k_rmsnorm(const float* __restrict__ x, const float* __restrict__ w,
                          float* __restrict__ xn) {
    int pos = blockIdx.x * blockDim.x + threadIdx.x;
    if (pos >= BL) return;
    int b = pos >> 10, p = pos & 1023;
    const float* xb = x + (size_t)b * C_ * HW_ + p;
    float s = 0.f;
    for (int c = 0; c < C_; ++c) { float v = xb[c * HW_]; s += v * v; }
    float inv = rsqrtf(s * (1.0f / C_) + 1e-6f);
    float* o = xn + (size_t)pos * C_;
    for (int c = 0; c < C_; ++c) o[c] = xb[c * HW_] * inv * w[c];
}

// ---- stage A tile (f32 -> bf16): 128x32 into padded LDS, 4 float4 per thread ----
__device__ __forceinline__ void stage_a_tile(const float* __restrict__ A, int K, int m0,
                                             int kk, int tid, u16b (*As)[40]) {
    #pragma unroll
    for (int i = 0; i < 4; ++i) {
        int q = tid + 256 * i;
        int row = q >> 3, col = (q & 7) * 4;
        const float4 f = *(const float4*)&A[(size_t)(m0 + row) * K + kk + col];
        As[row][col + 0] = f2bf(f.x);
        As[row][col + 1] = f2bf(f.y);
        As[row][col + 2] = f2bf(f.z);
        As[row][col + 3] = f2bf(f.w);
    }
}

// ---- TDM: async 64x32-elem bf16 tile (row stride K elems) -> LDS with padding ----
// pad_interval code 3 (16 DW = one 64B row) + pad_amount code 3 (4 DW = 16B)
// reproduces the 40-ushort (80B) padded LDS row stride used by the WMMA frag reads.
__device__ __forceinline__ void tdm_issue_b_tile(const u16b* __restrict__ src, int K,
                                                 u16b* lds_dst) {
#if HAVE_TDM
    unsigned lds_off = (unsigned)(unsigned long long)(void*)lds_dst; // LDS aperture: addr[31:0]
    unsigned long long ga = (unsigned long long)(const void*)src;
    v4u g0;
    g0[0] = 1u;                                           // count=1, user descriptor
    g0[1] = lds_off;                                      // lds_addr
    g0[2] = (unsigned)(ga & 0xFFFFFFFFu);                 // global_addr[31:0]
    g0[3] = (unsigned)((ga >> 32) & 0x01FFFFFFu) | (2u << 30); // global_addr[56:32] | type=2
    v8i g1;
    g1[0] = (int)((1u << 16) | (1u << 20) | (3u << 22) | (3u << 25)); // data_size=2B, pad_en, intv, amt
    g1[1] = 0;                                            // atomic_barrier_addr=0, tdim0 lo16=0
    g1[2] = 0x0010;                                       // tdim0 hi16 (1M), tdim1 lo16=0
    g1[3] = (int)((32u << 16) | 0x0010u);                 // tdim1 hi16 (1M), tile_dim0=32
    g1[4] = 64;                                           // tile_dim1=64, tile_dim2=0
    g1[5] = K;                                            // tensor_dim0_stride lo32
    g1[6] = 0;
    g1[7] = 0;
    v4i z4 = {0, 0, 0, 0};
#if defined(__clang_major__) && __clang_major__ >= 23
    v8i z8 = {0, 0, 0, 0, 0, 0, 0, 0};
    __builtin_amdgcn_tensor_load_to_lds(g0, g1, z4, z4, z8, 0);
#else
    __builtin_amdgcn_tensor_load_to_lds(g0, g1, z4, z4, 0);
#endif
#endif
}

// ---- WMMA bf16 GEMM: D[M][N] = alpha * A[M][K] (f32, cvt on stage) * Bw[N][K]^T ----
// 256 thr (8 waves), block tile 128x64, wave tile 32x32, K-step 32,
// double-buffered LDS; B tiles via TDM async DMA, A tiles VALU-converted.
__global__ __launch_bounds__(256) void k_gemm_bf16(
    const float* __restrict__ A, const u16b* __restrict__ Bw, float* __restrict__ Dm,
    int M, int N, int K, const float* __restrict__ alpha_ptr) {
    __shared__ u16b As[2][128][40];   // 80B row stride: conflict-free frag reads
    __shared__ u16b Bs[2][64][40];
    const int tid  = threadIdx.x;
    const int m0   = blockIdx.y * 128;
    const int n0   = blockIdx.x * 64;
    const int wave = tid >> 5, lane = tid & 31;
    const int wm = wave & 3, wn = wave >> 2;
    const int lr = lane & 15;
    const bool hi = (lane >= 16);
    const int NK = K >> 5;
    v8f acc00 = {}, acc01 = {}, acc10 = {}, acc11 = {};
    union FrU { uint4 q[2]; v16bf v; };

    // prologue: fill buffer 0
    stage_a_tile(A, K, m0, 0, tid, As[0]);
#if HAVE_TDM
    if (wave == 0) {
        tdm_issue_b_tile(&Bw[(size_t)n0 * K], K, &Bs[0][0][0]);
        __builtin_amdgcn_s_wait_tensorcnt(0);
    }
#else
    { int row = tid >> 2, col = (tid & 3) * 8;
      *(uint4*)&Bs[0][row][col] = *(const uint4*)&Bw[(size_t)(n0 + row) * K + col]; }
#endif
    __syncthreads();

    for (int ks = 0; ks < NK; ++ks) {
        const int cur = ks & 1, nxt = cur ^ 1;
        const bool more = (ks + 1 < NK);
        if (more) {
            stage_a_tile(A, K, m0, (ks + 1) << 5, tid, As[nxt]);
#if HAVE_TDM
            if (wave == 0)
                tdm_issue_b_tile(&Bw[(size_t)n0 * K + ((ks + 1) << 5)], K, &Bs[nxt][0][0]);
#else
            { int row = tid >> 2, col = (tid & 3) * 8;
              *(uint4*)&Bs[nxt][row][col] =
                  *(const uint4*)&Bw[(size_t)(n0 + row) * K + ((ks + 1) << 5) + col]; }
#endif
        }
        // A frag: row M=lr, K {0..7,16..23} (lo lanes) / {8..15,24..31} (hi lanes)
        FrU a0, a1, b0, b1;
        const int akb = hi ? 8 : 0;
        a0.q[0] = *(const uint4*)&As[cur][wm * 32 + lr][akb];
        a0.q[1] = *(const uint4*)&As[cur][wm * 32 + lr][akb + 16];
        a1.q[0] = *(const uint4*)&As[cur][wm * 32 + 16 + lr][akb];
        a1.q[1] = *(const uint4*)&As[cur][wm * 32 + 16 + lr][akb + 16];
        // B frag: col N=lr, K {0..7,16..23} (lo lanes) / {8..15,24..31} (hi lanes)
        const int bkb = hi ? 8 : 0;
        b0.q[0] = *(const uint4*)&Bs[cur][wn * 32 + lr][bkb];
        b0.q[1] = *(const uint4*)&Bs[cur][wn * 32 + lr][bkb + 16];
        b1.q[0] = *(const uint4*)&Bs[cur][wn * 32 + 16 + lr][bkb];
        b1.q[1] = *(const uint4*)&Bs[cur][wn * 32 + 16 + lr][bkb + 16];
        acc00 = __builtin_amdgcn_wmma_f32_16x16x32_bf16(false, a0.v, false, b0.v, (short)0, acc00, false, false);
        acc01 = __builtin_amdgcn_wmma_f32_16x16x32_bf16(false, a0.v, false, b1.v, (short)0, acc01, false, false);
        acc10 = __builtin_amdgcn_wmma_f32_16x16x32_bf16(false, a1.v, false, b0.v, (short)0, acc10, false, false);
        acc11 = __builtin_amdgcn_wmma_f32_16x16x32_bf16(false, a1.v, false, b1.v, (short)0, acc11, false, false);
#if HAVE_TDM
        if (wave == 0 && more) __builtin_amdgcn_s_wait_tensorcnt(0);
#endif
        __syncthreads();
    }
    float alpha = alpha_ptr ? alpha_ptr[0] : 1.0f;
    const int rb = m0 + wm * 32 + (hi ? 8 : 0);
    const int cb = n0 + wn * 32 + lr;
    #pragma unroll
    for (int r = 0; r < 8; ++r) {
        Dm[(size_t)(rb + r) * N + cb]           = acc00[r] * alpha;
        Dm[(size_t)(rb + r) * N + cb + 16]      = acc01[r] * alpha;
        Dm[(size_t)(rb + 16 + r) * N + cb]      = acc10[r] * alpha;
        Dm[(size_t)(rb + 16 + r) * N + cb + 16] = acc11[r] * alpha;
    }
}

// ---- causal depthwise conv1d (taps 4) + SiLU; dir=1 flips the sequence ----
__global__ void k_conv_silu(const float* __restrict__ xz, const float* __restrict__ cw,
                            const float* __restrict__ cb, float* __restrict__ u, int dir) {
    int i = blockIdx.x * blockDim.x + threadIdx.x;
    if (i >= BL * C_) return;
    int c = i % C_, bt = i / C_;
    int b = bt >> 10, t = bt & 1023;
    float acc = cb[c];
    #pragma unroll
    for (int j = 0; j < 4; ++j) {
        int tau = t - 3 + j;
        if (tau < 0) continue;
        int p = dir ? (1023 - tau) : tau;
        acc += cw[c * 4 + j] * xz[(size_t)(b * 1024 + p) * 768 + c];
    }
    u[i] = silu_(acc);
}

// ---- fused dt_proj + softplus + selective scan; one thread per (b, c) ----
__global__ __launch_bounds__(128) void k_scan(
    const float* __restrict__ xdbl, const float* __restrict__ u,
    const float* __restrict__ dtw, const float* __restrict__ dtb,
    const float* __restrict__ A_log, float* __restrict__ ys) {
    __shared__ float lrow[8 * 64];          // 8 timesteps of x_dbl rows
    int b = blockIdx.y;
    int c = blockIdx.x * 128 + threadIdx.x;
    float wdt[DTRANK];
    #pragma unroll
    for (int r = 0; r < DTRANK; ++r) wdt[r] = dtw[c * DTRANK + r];
    float bias = dtb[c];
    float Ar[NSTATE], h[NSTATE];
    #pragma unroll
    for (int n = 0; n < NSTATE; ++n) { Ar[n] = -__expf(A_log[c * NSTATE + n]); h[n] = 0.f; }
    for (int tc = 0; tc < 128; ++tc) {
        __syncthreads();
        #pragma unroll
        for (int i = 0; i < 4; ++i) {
            int e = threadIdx.x + 128 * i;
            lrow[e] = xdbl[(size_t)(b * 1024 + tc * 8 + (e >> 6)) * 64 + (e & 63)];
        }
        __syncthreads();
        for (int t8 = 0; t8 < 8; ++t8) {
            const float* row = &lrow[t8 * 64];
            int t = tc * 8 + t8;
            float d = bias;
            #pragma unroll
            for (int r = 0; r < DTRANK; ++r) d += wdt[r] * row[r];
            d = softplus_(d);
            float du  = d * u[(size_t)(b * 1024 + t) * C_ + c];
            float acc = 0.f;
            #pragma unroll
            for (int n = 0; n < NSTATE; ++n) {
                float dA = __expf(d * Ar[n]);
                h[n] = dA * h[n] + du * row[24 + n];   // Bm
                acc += h[n] * row[40 + n];             // Cm
            }
            ys[(size_t)(b * 1024 + t) * C_ + c] = acc;
        }
    }
}

// ---- combine fwd/bwd scans, skip connection u*D, gate by silu(z) ----
__global__ void k_combine(const float* __restrict__ ys_f, const float* __restrict__ ys_b,
                          const float* __restrict__ u_f, const float* __restrict__ u_b,
                          const float* __restrict__ xz, const float* __restrict__ Dp,
                          float* __restrict__ yc) {
    int i = blockIdx.x * blockDim.x + threadIdx.x;
    if (i >= BL * C_) return;
    int c = i % C_, bt = i / C_;
    int b = bt >> 10, p = bt & 1023;
    size_t ir = (size_t)(b * 1024 + 1023 - p) * C_ + c;
    float Dc = Dp[c];
    float z  = xz[(size_t)bt * 768 + 384 + c];
    float y  = (ys_f[i] + u_f[i] * Dc) + (ys_b[ir] + u_b[ir] * Dc);
    yc[i] = y * silu_(z);
}

// ---- 3x3 depthwise sigmoid gate + residual, write NCHW ----
__global__ void k_final(const float* __restrict__ x, const float* __restrict__ xn,
                        const float* __restrict__ g, const float* __restrict__ dww,
                        const float* __restrict__ dwb, const float* __restrict__ res_scale,
                        float* __restrict__ out) {
    int i = blockIdx.x * blockDim.x + threadIdx.x;
    if (i >= B_ * C_ * HW_) return;
    int p = i & 1023, bc = i >> 10;
    int c = bc % C_, b = bc / C_;
    int hh = p >> 5, ww = p & 31;
    float s = dwb[c];
    #pragma unroll
    for (int dy = -1; dy <= 1; ++dy) {
        int y2 = hh + dy; if (y2 < 0 || y2 > 31) continue;
        #pragma unroll
        for (int dx = -1; dx <= 1; ++dx) {
            int x2 = ww + dx; if (x2 < 0 || x2 > 31) continue;
            s += dww[c * 9 + (dy + 1) * 3 + (dx + 1)] *
                 xn[(size_t)(b * 1024 + y2 * 32 + x2) * C_ + c];
        }
    }
    float xnv   = xn[(size_t)(b * 1024 + p) * C_ + c];
    float local = sigm_(s) * xnv;
    out[i] = x[i] + res_scale[0] * (g[(size_t)(b * 1024 + p) * C_ + c] + local);
}

extern "C" void kernel_launch(void* const* d_in, const int* in_sizes, int n_in,
                              void* d_out, int out_size, void* d_ws, size_t ws_size,
                              hipStream_t stream) {
    (void)in_sizes; (void)n_in; (void)out_size; (void)ws_size;
    const float* x         = (const float*)d_in[0];
    const float* norm_w    = (const float*)d_in[1];
    const float* in_proj_w = (const float*)d_in[2];
    const float* conv1d_w  = (const float*)d_in[3];
    const float* conv1d_b  = (const float*)d_in[4];
    const float* x_proj_w  = (const float*)d_in[5];
    const float* dt_proj_w = (const float*)d_in[6];
    const float* dt_proj_b = (const float*)d_in[7];
    const float* A_log     = (const float*)d_in[8];
    const float* D_param   = (const float*)d_in[9];
    const float* out_proj_w= (const float*)d_in[10];
    const float* scale_mod = (const float*)d_in[11];
    const float* dw_w      = (const float*)d_in[12];
    const float* dw_b      = (const float*)d_in[13];
    const float* res_scale = (const float*)d_in[14];
    float* out = (float*)d_out;

    char* wsb = (char*)d_ws;
    size_t off = 0;
    auto alloc = [&](size_t bytes) -> char* {
        char* p = wsb + off;
        off += (bytes + 255) & ~(size_t)255;
        return p;
    };
    float* xn    = (float*)alloc((size_t)BL * C_ * 4);
    u16b*  w_in  = (u16b*) alloc((size_t)768 * 384 * 2);
    u16b*  w_xp  = (u16b*) alloc((size_t)64 * 384 * 2);
    u16b*  w_out = (u16b*) alloc((size_t)384 * 384 * 2);
    float* xz    = (float*)alloc((size_t)BL * 768 * 4);
    float* u_f   = (float*)alloc((size_t)BL * C_ * 4);
    float* u_b   = (float*)alloc((size_t)BL * C_ * 4);
    float* xd_f  = (float*)alloc((size_t)BL * 64 * 4);
    float* xd_b  = (float*)alloc((size_t)BL * 64 * 4);
    float* ys_f  = (float*)alloc((size_t)BL * C_ * 4);
    float* ys_b  = (float*)alloc((size_t)BL * C_ * 4);
    float* y_c   = (float*)alloc((size_t)BL * C_ * 4);
    float* g     = (float*)alloc((size_t)BL * C_ * 4);

    // pack weights to bf16 (x_proj zero-padded 56 -> 64 rows: no GEMM bounds checks)
    k_pack_w<<<(768 * 384 + 255) / 256, 256, 0, stream>>>(in_proj_w, w_in, 768, 768, 384);
    k_pack_w<<<(64 * 384 + 255) / 256, 256, 0, stream>>>(x_proj_w, w_xp, 56, 64, 384);
    k_pack_w<<<(384 * 384 + 255) / 256, 256, 0, stream>>>(out_proj_w, w_out, 384, 384, 384);

    k_rmsnorm<<<BL / 256, 256, 0, stream>>>(x, norm_w, xn);

    // in_proj: [8192,384] x [384,768] (shared by both directions)
    k_gemm_bf16<<<dim3(768 / 64, BL / 128), 256, 0, stream>>>(xn, w_in, xz, BL, 768, 384, nullptr);

    int nbc = (BL * C_ + 255) / 256;
    k_conv_silu<<<nbc, 256, 0, stream>>>(xz, conv1d_w, conv1d_b, u_f, 0);
    k_conv_silu<<<nbc, 256, 0, stream>>>(xz, conv1d_w, conv1d_b, u_b, 1);

    // x_proj: [8192,384] x [384,64(padded)]
    k_gemm_bf16<<<dim3(1, BL / 128), 256, 0, stream>>>(u_f, w_xp, xd_f, BL, 64, 384, nullptr);
    k_gemm_bf16<<<dim3(1, BL / 128), 256, 0, stream>>>(u_b, w_xp, xd_b, BL, 64, 384, nullptr);

    // selective scan (fused dt_proj + softplus), per direction
    k_scan<<<dim3(3, 8), 128, 0, stream>>>(xd_f, u_f, dt_proj_w, dt_proj_b, A_log, ys_f);
    k_scan<<<dim3(3, 8), 128, 0, stream>>>(xd_b, u_b, dt_proj_w, dt_proj_b, A_log, ys_b);

    k_combine<<<nbc, 256, 0, stream>>>(ys_f, ys_b, u_f, u_b, xz, D_param, y_c);

    // out_proj with alpha = scale_mod (out_proj(f)+out_proj(b) = out_proj(f+b))
    k_gemm_bf16<<<dim3(384 / 64, BL / 128), 256, 0, stream>>>(y_c, w_out, g, BL, 384, 384, scale_mod);

    k_final<<<(B_ * C_ * HW_ + 255) / 256, 256, 0, stream>>>(x, xn, g, dw_w, dw_b, res_scale, out);
}
